// DGCNN_47931835023374
// MI455X (gfx1250) — matrix operand order repaired and checked by reading, not compiled, not run on hardware
//
#include <hip/hip_runtime.h>

typedef __attribute__((ext_vector_type(16))) _Float16 v16h;
typedef __attribute__((ext_vector_type(8)))  _Float16 v8h;
typedef __attribute__((ext_vector_type(8)))  float    v8f;

#define NB   16
#define NPTS 2048
#define NBN  (NB * NPTS)

__device__ __forceinline__ float relu_f(float v) { return v > 0.f ? v : 0.f; }

// ---- WMMA fragment loaders (CDNA5 16-bit layouts, wave32) -------------------
// A matrix 16x32 f16, row-major source. ISA 7.12.2: lane holds row m = lane%16,
// K = {half*8 .. half*8+7} (VGPR 0-3) and {16+half*8 ..} (VGPR 4-7), i.e. two
// contiguous 16-byte spans -> two 128-bit loads per lane.
__device__ __forceinline__ v16h loadA_f16(const _Float16* buf, int stride,
                                          int kbase, int lane) {
  const int m = lane & 15, half = lane >> 4;
  const _Float16* p = buf + (size_t)m * stride + kbase + half * 8;
  const v8h lo = *(const v8h*)p;
  const v8h hi = *(const v8h*)(p + 16);
  return __builtin_shufflevector(lo, hi, 0, 1, 2, 3, 4, 5, 6, 7,
                                 8, 9, 10, 11, 12, 13, 14, 15);
}

// Pre-packed B fragment: frag-major [frag][lane][16] f16, one v16h per lane.
__device__ __forceinline__ v16h loadB_pk(const _Float16* base, int frag,
                                         int lane) {
  return *(const v16h*)(base + ((size_t)frag * 32 + lane) * 16);
}

// ---- Kernel 0: pack fp32 weights W[out][in] into B-fragment f16 layout ------
// B 32x16: lane -> col N = lane%16; VGPR v holds K = half*8 + 2v (+1) for v<4, 16 + half*8 + 2(v-4) (+1) for v>=4.
__global__ __launch_bounds__(128) void pack_b_kernel(
    const float* __restrict__ W, int ld, int klim, int nOT, int nKC,
    _Float16* __restrict__ dst) {
  const int t = blockIdx.x * 128 + threadIdx.x;   // one thread per (frag,lane)
  if (t >= nOT * nKC * 32) return;
  const int lane = t & 31, frag = t >> 5;
  const int kc = frag % nKC, ot = frag / nKC;
  const int ncol = lane & 15, half = lane >> 4;
  const float* wrow = W + (size_t)(ot * 16 + ncol) * ld;
  v16h r;
  #pragma unroll
  for (int v = 0; v < 8; ++v) {
    const int k = (v < 4 ? half * 8 + 2 * v : 16 + half * 8 + 2 * (v - 4)) + kc * 32;
    r[2 * v]     = (k     < klim) ? (_Float16)wrow[k]     : (_Float16)0.f;
    r[2 * v + 1] = (k + 1 < klim) ? (_Float16)wrow[k + 1] : (_Float16)0.f;
  }
  *(v16h*)(dst + (size_t)t * 16) = r;
}

// ---- Kernel 1: liner Conv1d(3->8)+BN+ReLU, store point-major h[(b*N+n)*8+c] --
__global__ __launch_bounds__(256) void liner_kernel(
    const float* __restrict__ x, const float* __restrict__ w,
    const float* __restrict__ s, const float* __restrict__ bb,
    float* __restrict__ h) {
  const int p = blockIdx.x * 256 + threadIdx.x;   // 0..B*N-1
  const int b = p >> 11, n = p & (NPTS - 1);
  const float x0 = x[((size_t)b * 3 + 0) * NPTS + n];
  const float x1 = x[((size_t)b * 3 + 1) * NPTS + n];
  const float x2 = x[((size_t)b * 3 + 2) * NPTS + n];
  #pragma unroll
  for (int c = 0; c < 8; ++c) {
    float v = w[c * 3 + 0] * x0 + w[c * 3 + 1] * x1 + w[c * 3 + 2] * x2;
    h[(size_t)p * 8 + c] = relu_f(v * s[c] + bb[c]);
  }
}

// ---- Kernel 2: kNN top-16 (neg squared dist); batch cached in LDS as f16 ----
__global__ __launch_bounds__(256) void knn_kernel(const float* __restrict__ h,
                                                  int* __restrict__ idx) {
  __shared__ _Float16 sh[NPTS * 8];               // 32 KB of 320 KB WGP LDS
  const int b = blockIdx.x >> 3;
  const int qblk = blockIdx.x & 7;
  const float* hb = h + (size_t)b * NPTS * 8;
  for (int i = threadIdx.x; i < NPTS * 8; i += 256) sh[i] = (_Float16)hb[i];
  __syncthreads();

  const int q = qblk * 256 + threadIdx.x;
  float qv[8];
  #pragma unroll
  for (int c = 0; c < 8; ++c) qv[c] = (float)sh[q * 8 + c];

  float bd[16]; int bi[16];
  #pragma unroll
  for (int j = 0; j < 16; ++j) { bd[j] = -3.0e38f; bi[j] = 0; }

  for (int m = 0; m < NPTS; ++m) {
    float ss = 0.f;
    #pragma unroll
    for (int c = 0; c < 8; ++c) {
      const float t = qv[c] - (float)sh[m * 8 + c];
      ss += t * t;
    }
    const float d = -ss;                          // = 2*inner - xx_q - xx_m
    if (d > bd[15]) {                             // strict > keeps tie order
      bd[15] = d; bi[15] = m;
      #pragma unroll
      for (int j = 15; j > 0; --j) {              // one bubble pass: sorted
        if (bd[j] > bd[j - 1]) {
          const float td = bd[j]; bd[j] = bd[j - 1]; bd[j - 1] = td;
          const int   ti = bi[j]; bi[j] = bi[j - 1]; bi[j - 1] = ti;
        }
      }
    }
  }
  int* op = idx + (size_t)(b * NPTS + q) * 16;
  #pragma unroll
  for (int j = 0; j < 16; ++j) op[j] = bi[j];     // sorted desc; top-12 = first 12
}

// ---- Kernel 3: fused branch (edge feat -> conv1 -> conv2 -> att pool -> proj)
// One wave per point; 4 waves (128 thr) per block; all matmuls via WMMA with
// pre-packed f16 B fragments.
template <int KN>
__global__ __launch_bounds__(128) void branch_kernel(
    const float* __restrict__ h, const int* __restrict__ idx,
    const _Float16* __restrict__ pw1, const float* __restrict__ s1, const float* __restrict__ b1,
    const _Float16* __restrict__ pw2, const float* __restrict__ s2, const float* __restrict__ b2,
    const _Float16* __restrict__ patt, const float* __restrict__ aw,
    const float* __restrict__ as_, const float* __restrict__ ab,
    _Float16* __restrict__ fcat, int coff) {
  __shared__ _Float16 sE [4][16 * 32];    // edge features (K padded to 32)
  __shared__ _Float16 sG1[4][16 * 64];    // conv1 output
  __shared__ _Float16 sG2[4][16 * 128];   // conv2 output
  __shared__ float    sP [4][128];        // pooled

  const int lane = threadIdx.x & 31, wave = threadIdx.x >> 5;
  const int point = blockIdx.x * 4 + wave;
  const int b = point >> 11;
  const int half = lane >> 4, col16 = lane & 15;

  // Edge features: row k = [h_nbr - h_ctr | h_ctr], zero-pad rows >= KN, K 16..31
  if (lane < 16) {
    _Float16* e = &sE[wave][lane * 32];
    if (lane < KN) {
      const float* ctr = h + (size_t)point * 8;
      const int nb = idx[(size_t)point * 16 + lane];
      const float* nbr = h + (size_t)(b * NPTS + nb) * 8;
      #pragma unroll
      for (int c = 0; c < 8; ++c) {
        const float cv = ctr[c];
        e[c]     = (_Float16)(nbr[c] - cv);
        e[c + 8] = (_Float16)cv;
      }
    } else {
      #pragma unroll
      for (int c = 0; c < 16; ++c) e[c] = (_Float16)0.f;
    }
    #pragma unroll
    for (int c = 16; c < 32; ++c) e[c] = (_Float16)0.f;
  }
  __syncthreads();

  // Stage 1: (16x16) @ w1^T -> 16x64, ReLU(BN)   [w1: nOT=4, nKC=1]
  {
    const v16h a = loadA_f16(&sE[wave][0], 32, 0, lane);
    #pragma unroll
    for (int t = 0; t < 4; ++t) {
      v8f acc = {};
      acc = __builtin_amdgcn_wmma_f32_16x16x32_f16(
          false, a, false, loadB_pk(pw1, t, lane), (short)0, acc, false, false);
      const int c = t * 16 + col16;
      const float sc = s1[c], bi = b1[c];
      #pragma unroll
      for (int i = 0; i < 8; ++i)
        sG1[wave][(i + 8 * half) * 64 + c] = (_Float16)relu_f(acc[i] * sc + bi);
    }
  }
  __syncthreads();

  // Stage 2: (16x64) @ w2^T -> 16x128, ReLU(BN)  [w2: nOT=8, nKC=2]
  {
    const v16h a0 = loadA_f16(&sG1[wave][0], 64, 0,  lane);
    const v16h a1 = loadA_f16(&sG1[wave][0], 64, 32, lane);
    #pragma unroll
    for (int t = 0; t < 8; ++t) {
      v8f acc = {};
      acc = __builtin_amdgcn_wmma_f32_16x16x32_f16(
          false, a0, false, loadB_pk(pw2, t * 2 + 0, lane), (short)0, acc, false, false);
      acc = __builtin_amdgcn_wmma_f32_16x16x32_f16(
          false, a1, false, loadB_pk(pw2, t * 2 + 1, lane), (short)0, acc, false, false);
      const int c = t * 16 + col16;
      const float sc = s2[c], bi = b2[c];
      #pragma unroll
      for (int i = 0; i < 8; ++i)
        sG2[wave][(i + 8 * half) * 128 + c] = (_Float16)relu_f(acc[i] * sc + bi);
    }
  }
  __syncthreads();

  // Attention: sigmoid((16x128) @ att^T), gated sum over valid k rows
  // [att: nOT=8, nKC=4]
  {
    v16h ak[4];
    #pragma unroll
    for (int kc = 0; kc < 4; ++kc)
      ak[kc] = loadA_f16(&sG2[wave][0], 128, kc * 32, lane);
    #pragma unroll
    for (int t = 0; t < 8; ++t) {
      v8f acc = {};
      #pragma unroll
      for (int kc = 0; kc < 4; ++kc)
        acc = __builtin_amdgcn_wmma_f32_16x16x32_f16(
            false, ak[kc], false, loadB_pk(patt, t * 4 + kc, lane),
            (short)0, acc, false, false);
      const int c = t * 16 + col16;
      float p = 0.f;
      #pragma unroll
      for (int i = 0; i < 8; ++i) {
        const int row = i + 8 * half;
        if (row < KN) {                       // mask padded neighbor rows
          const float gate = 1.f / (1.f + __expf(-acc[i]));
          p += (float)sG2[wave][row * 128 + c] * gate;
        }
      }
      p += __shfl_xor(p, 16, 32);             // combine row halves
      if (half == 0) sP[wave][c] = p;
    }
  }
  __syncthreads();

  // Per-branch projection 128 -> 256 (matrix-vector on VALU), vector store
  {
    float accv[8];
    #pragma unroll
    for (int j = 0; j < 8; ++j) {
      const int o = lane * 8 + j;
      const float* wr = aw + (size_t)o * 128;
      float acc = 0.f;
      for (int i = 0; i < 128; ++i) acc += sP[wave][i] * wr[i];
      accv[j] = relu_f(acc * as_[o] + ab[o]);
    }
    v8h ov;
    #pragma unroll
    for (int j = 0; j < 8; ++j) ov[j] = (_Float16)accv[j];
    *(v8h*)(&fcat[(size_t)point * 512 + coff + lane * 8]) = ov;
  }
}

// ---- Kernel 4: final 512->512 GEMM + BN + ReLU, store (B,512,N) -------------
// One wave per (16-point x 16-out) tile, K=512 via 16 chained WMMAs.
__global__ __launch_bounds__(256) void final_kernel(
    const _Float16* __restrict__ f, const _Float16* __restrict__ pwf,
    const float* __restrict__ sf, const float* __restrict__ bf,
    float* __restrict__ out) {
  const int lane = threadIdx.x & 31, wave = threadIdx.x >> 5;
  const int tid = blockIdx.x * 8 + wave;      // 65536 tiles total
  const int mtile = tid >> 5, otile = tid & 31;
  const _Float16* fr = f + (size_t)mtile * 16 * 512;
  const int m = lane & 15, half = lane >> 4;

  v8f acc = {};
  for (int kc = 0; kc < 16; ++kc) {
    if (kc + 1 < 16)                          // gfx1250 global_prefetch_b8
      __builtin_prefetch(fr + (size_t)m * 512 + (kc + 1) * 32 + half * 8, 0, 1);
    const v16h a  = loadA_f16(fr, 512, kc * 32, lane);
    const v16h bw = loadB_pk(pwf, otile * 16 + kc, lane);
    acc = __builtin_amdgcn_wmma_f32_16x16x32_f16(false, a, false, bw,
                                                 (short)0, acc, false, false);
  }
  const int o = otile * 16 + m;
  const float sc = sf[o], bi = bf[o];
  #pragma unroll
  for (int i = 0; i < 8; ++i) {
    const int point = mtile * 16 + i + 8 * half;
    const int bb = point >> 11, nn = point & (NPTS - 1);
    out[((size_t)bb * 512 + o) * NPTS + nn] = relu_f(acc[i] * sc + bi);
  }
}

extern "C" void kernel_launch(void* const* d_in, const int* in_sizes, int n_in,
                              void* d_out, int out_size, void* d_ws, size_t ws_size,
                              hipStream_t stream) {
  (void)in_sizes; (void)n_in; (void)out_size; (void)ws_size;
  const float* x    = (const float*)d_in[0];
  const float* lw   = (const float*)d_in[1];
  const float* ls   = (const float*)d_in[2];
  const float* lb   = (const float*)d_in[3];
  const float* w1   = (const float*)d_in[4];
  const float* s1   = (const float*)d_in[5];
  const float* b1   = (const float*)d_in[6];
  const float* w2   = (const float*)d_in[7];
  const float* s2   = (const float*)d_in[8];
  const float* b2   = (const float*)d_in[9];
  const float* w3   = (const float*)d_in[10];
  const float* s3   = (const float*)d_in[11];
  const float* b3   = (const float*)d_in[12];
  const float* w4   = (const float*)d_in[13];
  const float* s4   = (const float*)d_in[14];
  const float* b4   = (const float*)d_in[15];
  const float* a1at = (const float*)d_in[16];
  const float* a1w  = (const float*)d_in[17];
  const float* a1s  = (const float*)d_in[18];
  const float* a1b  = (const float*)d_in[19];
  const float* a2at = (const float*)d_in[20];
  const float* a2w  = (const float*)d_in[21];
  const float* a2s  = (const float*)d_in[22];
  const float* a2b  = (const float*)d_in[23];
  const float* wf   = (const float*)d_in[24];
  const float* sf   = (const float*)d_in[25];
  const float* bf   = (const float*)d_in[26];

  // workspace layout (all sections 256B aligned)
  char* ws = (char*)d_ws;
  size_t off = 0;
  float*    h    = (float*)(ws + off);     off += (size_t)NBN * 8 * 4;    // 1 MB
  int*      idx  = (int*)(ws + off);       off += (size_t)NBN * 16 * 4;   // 2 MB
  _Float16* fcat = (_Float16*)(ws + off);  off += (size_t)NBN * 512 * 2;  // 32 MB
  _Float16* pw1  = (_Float16*)(ws + off);  off += (size_t)4  * 1  * 512 * 2;
  _Float16* pw2  = (_Float16*)(ws + off);  off += (size_t)8  * 2  * 512 * 2;
  _Float16* pat1 = (_Float16*)(ws + off);  off += (size_t)8  * 4  * 512 * 2;
  _Float16* pw3  = (_Float16*)(ws + off);  off += (size_t)4  * 1  * 512 * 2;
  _Float16* pw4  = (_Float16*)(ws + off);  off += (size_t)8  * 2  * 512 * 2;
  _Float16* pat2 = (_Float16*)(ws + off);  off += (size_t)8  * 4  * 512 * 2;
  _Float16* pwf  = (_Float16*)(ws + off);  off += (size_t)32 * 16 * 512 * 2;

  // pack all WMMA B operands once (frag = ot*nKC + kc; 32 lanes x 16 f16 each)
  pack_b_kernel<<<(4 * 1 * 32 + 127) / 128, 128, 0, stream>>>(w1, 16, 16, 4, 1, pw1);
  pack_b_kernel<<<(8 * 2 * 32 + 127) / 128, 128, 0, stream>>>(w2, 64, 64, 8, 2, pw2);
  pack_b_kernel<<<(8 * 4 * 32 + 127) / 128, 128, 0, stream>>>(a1at, 128, 128, 8, 4, pat1);
  pack_b_kernel<<<(4 * 1 * 32 + 127) / 128, 128, 0, stream>>>(w3, 16, 16, 4, 1, pw3);
  pack_b_kernel<<<(8 * 2 * 32 + 127) / 128, 128, 0, stream>>>(w4, 64, 64, 8, 2, pw4);
  pack_b_kernel<<<(8 * 4 * 32 + 127) / 128, 128, 0, stream>>>(a2at, 128, 128, 8, 4, pat2);
  pack_b_kernel<<<(32 * 16 * 32 + 127) / 128, 128, 0, stream>>>(wf, 512, 512, 32, 16, pwf);

  liner_kernel<<<NBN / 256, 256, 0, stream>>>(x, lw, ls, lb, h);
  knn_kernel<<<NB * (NPTS / 256), 256, 0, stream>>>(h, idx);
  branch_kernel<16><<<NBN / 4, 128, 0, stream>>>(
      h, idx, pw1, s1, b1, pw2, s2, b2, pat1, a1w, a1s, a1b, fcat, 0);
  branch_kernel<12><<<NBN / 4, 128, 0, stream>>>(
      h, idx, pw3, s3, b3, pw4, s4, b4, pat2, a2w, a2s, a2b, fcat, 256);
  final_kernel<<<(NBN / 16) * 32 / 8, 256, 0, stream>>>(fcat, pwf, sf, bf,
                                                        (float*)d_out);
}
